// StepWiseMLPAutoEncoder_46737834115440
// MI455X (gfx1250) — compile-verified, matrix operand order checked by reading, not run on hardware
//
#include <hip/hip_runtime.h>
#include <hip/hip_bf16.h>

// ---------------------------------------------------------------------------
// CDNA5 (gfx1250) StepWiseMLPAutoEncoder
//   encoder: 3 dense bf16 WMMA GEMMs over 16384 rows (LDS-staged B panels)
//   decoder: 256 sequential steps, 3 bf16 WMMA GEMMs each (M=64)
// ---------------------------------------------------------------------------

typedef __attribute__((ext_vector_type(16))) __bf16 v16bf;
typedef __attribute__((ext_vector_type(8)))  __bf16 v8bf;
typedef __attribute__((ext_vector_type(8)))  float  v8f;
typedef __attribute__((ext_vector_type(4)))  int    v4i;

// Problem constants
#define BATCH 64
#define SEQ   256
#define STFT  1024
// Padded encoder dims (K multiple of 64 for LDS chunking, N tiles multiple of 4)
#define EK1 2112   // 2049 -> 2112
#define EN1 960    // 944  -> 960  (60 tiles, /4 ok; also K of layer 2, /64 ok)
#define EK2 960
#define EN2 448    // 416 -> 448  (28 tiles)
#define EK3 448
#define EN3 32
// Padded decoder dims
#define DK1 1088   // 1057 -> 1088 (17 x 64)
#define DN1 1088   // 1046 -> 1088 (68 tiles)
#define DK2 1088
#define DN2 1088   // 1035 -> 1088
#define DK3 1088
#define DN3 1024
#define KC  64     // K chunk staged in LDS per iteration

#if defined(__has_builtin)
#if __has_builtin(__builtin_amdgcn_global_load_async_to_lds_b128)
#define HAVE_ASYNC_LDS 1
#endif
#endif

__device__ __forceinline__ __bf16 f2bf(float f) {
  unsigned u = __builtin_bit_cast(unsigned, f);
  u += 0x7FFFu + ((u >> 16) & 1u);              // round-to-nearest-even
  return __builtin_bit_cast(__bf16, (unsigned short)(u >> 16));
}

__device__ __forceinline__ v16bf pack16(v8bf lo, v8bf hi) {
  v16bf r;
#pragma unroll
  for (int i = 0; i < 8; ++i) { r[i] = lo[i]; r[i + 8] = hi[i]; }
  return r;
}

// Stage a 64-row x KC-col bf16 B panel chunk into LDS (16B per transfer).
// gB points at the block's first B row; source chunk begins at column kk.
__device__ __forceinline__ void stage_B(const __bf16* __restrict__ gB, int Kpad, int kk,
                                        __bf16* lB, int tid, int nthr)
{
  for (int q = tid; q < 512; q += nthr) {      // 64 rows * 8 chunks of 8 bf16
    const int r = q >> 3;
    const int c = (q & 7) << 3;
    const __bf16* gp = gB + (size_t)r * Kpad + kk + c;
    __bf16*       lp = lB + r * KC + c;
#if HAVE_ASYNC_LDS
    __builtin_amdgcn_global_load_async_to_lds_b128(
        (__attribute__((address_space(1))) v4i*)(size_t)gp,
        (__attribute__((address_space(3))) v4i*)(unsigned)(size_t)lp,
        0, 0);
#else
    *(v8bf*)lp = *(const v8bf*)gp;             // global_load_b128 + ds_store_b128
#endif
  }
}

__device__ __forceinline__ void async_fence() {
#if HAVE_ASYNC_LDS
  asm volatile("s_wait_asynccnt 0x0" ::: "memory");
#endif
}

// 8 WMMAs on one KC chunk: A fragment reused across 4 N-tiles from the LDS panel.
__device__ __forceinline__ void compute_chunk(const __bf16* lB, const __bf16* arow,
                                              int kk, int koff, int l16, v8f acc[4])
{
#pragma unroll
  for (int ks = 0; ks < KC; ks += 32) {
    const v8bf a0 = *(const v8bf*)(arow + kk + ks + koff);
    const v8bf a1 = *(const v8bf*)(arow + kk + ks + koff + 16);
    const v16bf av = pack16(a0, a1);
#pragma unroll
    for (int j = 0; j < 4; ++j) {
      const __bf16* bl = lB + (j * 16 + l16) * KC + ks + koff;   // ds_load_b128 x2
      const v8bf b0 = *(const v8bf*)bl;
      const v8bf b1 = *(const v8bf*)(bl + 16);
      acc[j] = __builtin_amdgcn_wmma_f32_16x16x32_bf16(
          false, av, false, pack16(b0, b1), (short)0, acc[j], false, false);
    }
  }
}

// ---------------------------------------------------------------------------
// Blocked bf16 WMMA GEMM: block = WPB waves; block tile (WPB*16) x 64.
// Wave w owns M-tile (blockIdx.y*WPB + w) and 4 N-tiles; B panel shared in LDS.
// ---------------------------------------------------------------------------
__global__ __launch_bounds__(256) void gemm_blk(
    const __bf16* __restrict__ A, const __bf16* __restrict__ Bt,
    const float* __restrict__ bias, __bf16* __restrict__ C,
    int Kpad, int ldc, int relu)
{
  __shared__ __bf16 Bs[2][64 * KC];
  const int tid  = threadIdx.x;
  const int nthr = blockDim.x;
  const int wave = tid >> 5, lane = tid & 31;
  const int l16  = lane & 15;
  const int koff = (lane >> 4) << 3;
  const int mtile = blockIdx.y * (nthr >> 5) + wave;
  const __bf16* arow  = A + (size_t)(mtile * 16 + l16) * Kpad;
  const __bf16* gBblk = Bt + (size_t)(blockIdx.x * 64) * Kpad;

  v8f acc[4] = {};
  stage_B(gBblk, Kpad, 0, &Bs[0][0], tid, nthr);
  async_fence();
  __syncthreads();
  int buf = 0;
  for (int kk = 0; kk < Kpad; kk += KC) {
    if (kk + KC < Kpad) stage_B(gBblk, Kpad, kk + KC, &Bs[buf ^ 1][0], tid, nthr);
    __builtin_prefetch(arow + kk + 2 * KC, 0, 3);
    compute_chunk(&Bs[buf][0], arow, kk, koff, l16, acc);
    async_fence();
    __syncthreads();
    buf ^= 1;
  }

  const int mbase = mtile * 16 + ((lane >> 4) << 3);
#pragma unroll
  for (int j = 0; j < 4; ++j) {
    const int n = (blockIdx.x * 4 + j) * 16 + l16;
    const float bb = bias[n];
#pragma unroll
    for (int i = 0; i < 8; ++i) {
      float v = acc[j][i] + bb;
      if (relu) v = v > 0.f ? v : 0.f;
      C[(size_t)(mbase + i) * ldc + n] = f2bf(v);
    }
  }
}

// ---------------------------------------------------------------------------
// Simple per-wave GEMM (kept for the tiny encoder layer 3: N = 32)
// ---------------------------------------------------------------------------
__global__ __launch_bounds__(256) void gemm_simple(
    const __bf16* __restrict__ A, const __bf16* __restrict__ Bt,
    const float* __restrict__ bias, __bf16* __restrict__ C,
    int Mtiles, int Ntiles, int Kpad, int ldc, int relu)
{
  const int wave = blockIdx.x * (blockDim.x >> 5) + (threadIdx.x >> 5);
  const int lane = threadIdx.x & 31;
  if (wave >= Mtiles * Ntiles) return;
  const int tm = wave / Ntiles, tn = wave % Ntiles;
  const int l16  = lane & 15;
  const int koff = (lane >> 4) << 3;
  const __bf16* arow = A  + (size_t)(tm * 16 + l16) * Kpad;
  const __bf16* brow = Bt + (size_t)(tn * 16 + l16) * Kpad;
  v8f acc = {};
  for (int kk = 0; kk < Kpad; kk += 32) {
    __builtin_prefetch(arow + kk + 128, 0, 3);
    __builtin_prefetch(brow + kk + 128, 0, 3);
    const v8bf a0 = *(const v8bf*)(arow + kk + koff);
    const v8bf a1 = *(const v8bf*)(arow + kk + koff + 16);
    const v8bf b0 = *(const v8bf*)(brow + kk + koff);
    const v8bf b1 = *(const v8bf*)(brow + kk + koff + 16);
    acc = __builtin_amdgcn_wmma_f32_16x16x32_bf16(
        false, pack16(a0, a1), false, pack16(b0, b1), (short)0, acc, false, false);
  }
  const int n     = tn * 16 + l16;
  const int mbase = tm * 16 + ((lane >> 4) << 3);
  const float bb  = bias[n];
#pragma unroll
  for (int i = 0; i < 8; ++i) {
    float v = acc[i] + bb;
    if (relu) v = v > 0.f ? v : 0.f;
    C[(size_t)(mbase + i) * ldc + n] = f2bf(v);
  }
}

// ---------------------------------------------------------------------------
// Decoder layer 1 (blocked): A assembled on the fly from [ctrl(32)|prev(1024)|t];
// 8-element chunks never straddle region boundaries. 4 waves, N-block of 64.
// ---------------------------------------------------------------------------
__device__ __forceinline__ v8bf dec1_chunk(const __bf16* crow, const __bf16* prow,
                                           __bf16 tb, int kb) {
  if (kb < 32)        return *(const v8bf*)(crow + kb);
  else if (kb < 1056) return *(const v8bf*)(prow + (kb - 32));
  v8bf r;
#pragma unroll
  for (int i = 0; i < 8; ++i) r[i] = __builtin_bit_cast(__bf16, (unsigned short)0);
  if (kb == 1056) r[0] = tb;
  return r;
}

__global__ __launch_bounds__(128) void dec_layer1(
    const __bf16* __restrict__ ctrl, const __bf16* __restrict__ prev,
    const __bf16* __restrict__ W1t, const float* __restrict__ b1,
    __bf16* __restrict__ H1, int t)
{
  __shared__ __bf16 Bs[2][64 * KC];
  const int tid  = threadIdx.x;
  const int nthr = blockDim.x;
  const int wave = tid >> 5, lane = tid & 31;
  const int l16  = lane & 15;
  const int koff = (lane >> 4) << 3;
  const int b = wave * 16 + l16;                           // batch row (M=64)
  const __bf16* crow  = ctrl + ((size_t)b * SEQ + t) * 32;
  const __bf16* prow  = prev + (size_t)b * STFT;
  const __bf16* gBblk = W1t + (size_t)(blockIdx.x * 64) * DK1;
  const __bf16  tb    = f2bf((float)t * (1.0f / (float)SEQ));

  v8f acc[4] = {};
  stage_B(gBblk, DK1, 0, &Bs[0][0], tid, nthr);
  async_fence();
  __syncthreads();
  int buf = 0;
  for (int kk = 0; kk < DK1; kk += KC) {
    if (kk + KC < DK1) stage_B(gBblk, DK1, kk + KC, &Bs[buf ^ 1][0], tid, nthr);
#pragma unroll
    for (int ks = 0; ks < KC; ks += 32) {
      const v8bf a0 = dec1_chunk(crow, prow, tb, kk + ks + koff);
      const v8bf a1 = dec1_chunk(crow, prow, tb, kk + ks + koff + 16);
      const v16bf av = pack16(a0, a1);
#pragma unroll
      for (int j = 0; j < 4; ++j) {
        const __bf16* bl = &Bs[buf][0] + (j * 16 + l16) * KC + ks + koff;
        const v8bf b0 = *(const v8bf*)bl;
        const v8bf b1 = *(const v8bf*)(bl + 16);
        acc[j] = __builtin_amdgcn_wmma_f32_16x16x32_bf16(
            false, av, false, pack16(b0, b1), (short)0, acc[j], false, false);
      }
    }
    async_fence();
    __syncthreads();
    buf ^= 1;
  }

  const int mbase = wave * 16 + ((lane >> 4) << 3);
#pragma unroll
  for (int j = 0; j < 4; ++j) {
    const int n = (blockIdx.x * 4 + j) * 16 + l16;
    const float bb = b1[n];
#pragma unroll
    for (int i = 0; i < 8; ++i) {
      float v = acc[j][i] + bb;
      v = v > 0.f ? v : 0.f;
      H1[(size_t)(mbase + i) * DN1 + n] = f2bf(v);
    }
  }
}

// ---------------------------------------------------------------------------
// Decoder layer 3 (blocked): fp32 to d_out[b,s,t] and bf16 into feedback buffer.
// ---------------------------------------------------------------------------
__global__ __launch_bounds__(128) void dec_layer3(
    const __bf16* __restrict__ H2, const __bf16* __restrict__ W3t,
    const float* __restrict__ b3, float* __restrict__ out,
    __bf16* __restrict__ prevbf, int t)
{
  __shared__ __bf16 Bs[2][64 * KC];
  const int tid  = threadIdx.x;
  const int nthr = blockDim.x;
  const int wave = tid >> 5, lane = tid & 31;
  const int l16  = lane & 15;
  const int koff = (lane >> 4) << 3;
  const __bf16* arow  = H2 + (size_t)(wave * 16 + l16) * DK3;
  const __bf16* gBblk = W3t + (size_t)(blockIdx.x * 64) * DK3;

  v8f acc[4] = {};
  stage_B(gBblk, DK3, 0, &Bs[0][0], tid, nthr);
  async_fence();
  __syncthreads();
  int buf = 0;
  for (int kk = 0; kk < DK3; kk += KC) {
    if (kk + KC < DK3) stage_B(gBblk, DK3, kk + KC, &Bs[buf ^ 1][0], tid, nthr);
    compute_chunk(&Bs[buf][0], arow, kk, koff, l16, acc);
    async_fence();
    __syncthreads();
    buf ^= 1;
  }

  const int mbase = wave * 16 + ((lane >> 4) << 3);
#pragma unroll
  for (int j = 0; j < 4; ++j) {
    const int n = (blockIdx.x * 4 + j) * 16 + l16;
    const float bb = b3[n];
#pragma unroll
    for (int i = 0; i < 8; ++i) {
      float v = acc[j][i] + bb;                 // final layer: linear
      out[(size_t)(mbase + i) * (STFT * SEQ) + (size_t)n * SEQ + t] = v;
      prevbf[(size_t)(mbase + i) * STFT + n] = f2bf(v);
    }
  }
}

// ---------------------------------------------------------------------------
// Prep kernels
// ---------------------------------------------------------------------------
__global__ __launch_bounds__(256) void cvt_w(const float* __restrict__ W,
                                             __bf16* __restrict__ Wt,
                                             int K, int N, int Kpad, int Npad)
{
  long idx = (long)blockIdx.x * 256 + threadIdx.x;
  if (idx >= (long)Kpad * Npad) return;
  int k = (int)(idx % Kpad);
  int n = (int)(idx / Kpad);
  float v = (k < K && n < N) ? W[(size_t)k * N + n] : 0.f;
  Wt[idx] = f2bf(v);
}

__global__ __launch_bounds__(256) void cvt_b(const float* __restrict__ b,
                                             float* __restrict__ bp, int N, int Npad)
{
  int i = blockIdx.x * 256 + threadIdx.x;
  if (i < Npad) bp[i] = (i < N) ? b[i] : 0.f;
}

__global__ __launch_bounds__(256) void build_encA(const float* __restrict__ x,
                                                  __bf16* __restrict__ A)
{
  long idx = (long)blockIdx.x * 256 + threadIdx.x;
  const long total = (long)BATCH * SEQ * EK1;
  if (idx >= total) return;
  int k = (int)(idx % EK1);
  int r = (int)(idx / EK1);
  int b = r >> 8;            // r / SEQ
  int t = r & 255;           // r % SEQ
  float v = 0.f;
  if (k < STFT)            { if (t > 0) v = x[(size_t)b * STFT * SEQ + (size_t)k * SEQ + (t - 1)]; }
  else if (k < 2 * STFT)   v = x[(size_t)b * STFT * SEQ + (size_t)(k - STFT) * SEQ + t];
  else if (k == 2 * STFT)  v = (float)t * (1.0f / (float)SEQ);
  A[idx] = f2bf(v);
}

__global__ __launch_bounds__(256) void zero_bf(__bf16* __restrict__ p, int n)
{
  int i = blockIdx.x * 256 + threadIdx.x;
  if (i < n) p[i] = __builtin_bit_cast(__bf16, (unsigned short)0);
}

// ---------------------------------------------------------------------------
// Host launcher
// ---------------------------------------------------------------------------
static inline int nb(long n) { return (int)((n + 255) / 256); }

extern "C" void kernel_launch(void* const* d_in, const int* in_sizes, int n_in,
                              void* d_out, int out_size, void* d_ws, size_t ws_size,
                              hipStream_t stream) {
  const float* x   = (const float*)d_in[0];
  const float* eW1 = (const float*)d_in[1];
  const float* eb1 = (const float*)d_in[2];
  const float* eW2 = (const float*)d_in[3];
  const float* eb2 = (const float*)d_in[4];
  const float* eW3 = (const float*)d_in[5];
  const float* eb3 = (const float*)d_in[6];
  const float* dW1 = (const float*)d_in[7];
  const float* db1 = (const float*)d_in[8];
  const float* dW2 = (const float*)d_in[9];
  const float* db2 = (const float*)d_in[10];
  const float* dW3 = (const float*)d_in[11];
  const float* db3 = (const float*)d_in[12];
  float* out = (float*)d_out;

  char* ws = (char*)d_ws;
  size_t cur = 0;
  auto take = [&](size_t bytes) -> char* {
    char* p = ws + cur;
    cur = (cur + bytes + 255) & ~(size_t)255;
    return p;
  };
  const long ROWS = (long)BATCH * SEQ;   // 16384

  __bf16* encA   = (__bf16*)take(ROWS * EK1 * 2);
  __bf16* eW1t   = (__bf16*)take((size_t)EN1 * EK1 * 2);
  __bf16* eW2t   = (__bf16*)take((size_t)EN2 * EK2 * 2);
  __bf16* eW3t   = (__bf16*)take((size_t)EN3 * EK3 * 2);
  __bf16* dW1t   = (__bf16*)take((size_t)DN1 * DK1 * 2);
  __bf16* dW2t   = (__bf16*)take((size_t)DN2 * DK2 * 2);
  __bf16* dW3t   = (__bf16*)take((size_t)DN3 * DK3 * 2);
  __bf16* encH1  = (__bf16*)take(ROWS * EN1 * 2);
  __bf16* encH2  = (__bf16*)take(ROWS * EN2 * 2);
  __bf16* ctrl   = (__bf16*)take(ROWS * EN3 * 2);
  __bf16* prevbf = (__bf16*)take((size_t)BATCH * STFT * 2);
  __bf16* decH1  = (__bf16*)take((size_t)BATCH * DN1 * 2);
  __bf16* decH2  = (__bf16*)take((size_t)BATCH * DN2 * 2);
  float*  eb1p   = (float*)take(EN1 * 4);
  float*  eb2p   = (float*)take(EN2 * 4);
  float*  eb3p   = (float*)take(EN3 * 4);
  float*  db1p   = (float*)take(DN1 * 4);
  float*  db2p   = (float*)take(DN2 * 4);
  float*  db3p   = (float*)take(DN3 * 4);
  (void)in_sizes; (void)n_in; (void)out_size; (void)ws_size;

  // ---- weight / bias prep (bf16, transposed, zero-padded) ----
  cvt_w<<<nb((long)EN1 * EK1), 256, 0, stream>>>(eW1, eW1t, 2049,  944, EK1, EN1);
  cvt_w<<<nb((long)EN2 * EK2), 256, 0, stream>>>(eW2, eW2t,  944,  416, EK2, EN2);
  cvt_w<<<nb((long)EN3 * EK3), 256, 0, stream>>>(eW3, eW3t,  416,   32, EK3, EN3);
  cvt_w<<<nb((long)DN1 * DK1), 256, 0, stream>>>(dW1, dW1t, 1057, 1046, DK1, DN1);
  cvt_w<<<nb((long)DN2 * DK2), 256, 0, stream>>>(dW2, dW2t, 1046, 1035, DK2, DN2);
  cvt_w<<<nb((long)DN3 * DK3), 256, 0, stream>>>(dW3, dW3t, 1035, 1024, DK3, DN3);
  cvt_b<<<nb(EN1), 256, 0, stream>>>(eb1, eb1p,  944, EN1);
  cvt_b<<<nb(EN2), 256, 0, stream>>>(eb2, eb2p,  416, EN2);
  cvt_b<<<nb(EN3), 256, 0, stream>>>(eb3, eb3p,   32, EN3);
  cvt_b<<<nb(DN1), 256, 0, stream>>>(db1, db1p, 1046, DN1);
  cvt_b<<<nb(DN2), 256, 0, stream>>>(db2, db2p, 1035, DN2);
  cvt_b<<<nb(DN3), 256, 0, stream>>>(db3, db3p, 1024, DN3);

  // ---- encoder ----
  build_encA<<<nb(ROWS * EK1), 256, 0, stream>>>(x, encA);
  // L1: 1024 M-tiles (y=128 blocks of 8 waves) x 60 N-tiles (x=15 blocks of 4)
  gemm_blk<<<dim3(15, 128), 256, 0, stream>>>(encA,  eW1t, eb1p, encH1, EK1, EN1, 1);
  // L2: 1024 x 28 tiles
  gemm_blk<<<dim3(7, 128), 256, 0, stream>>>(encH1, eW2t, eb2p, encH2, EK2, EN2, 1);
  // L3 (linear, tiny N): 1024 x 2 tiles -> controls
  gemm_simple<<<2048 / 8, 256, 0, stream>>>(encH2, eW3t, eb3p, ctrl, 1024, EN3 / 16, EK3, EN3, 0);

  // ---- decoder scan ----
  zero_bf<<<nb((long)BATCH * STFT), 256, 0, stream>>>(prevbf, BATCH * STFT);
  for (int t = 0; t < SEQ; ++t) {
    dec_layer1<<<dim3(DN1 / 64, 1), 128, 0, stream>>>(ctrl, prevbf, dW1t, db1p, decH1, t);
    gemm_blk  <<<dim3(DN2 / 64, 1), 128, 0, stream>>>(decH1, dW2t, db2p, decH2, DK2, DN2, 1);
    dec_layer3<<<dim3(DN3 / 64, 1), 128, 0, stream>>>(decH2, dW3t, db3p, out, prevbf, t);
  }
}
